// Trajectron_87797721465263
// MI455X (gfx1250) — matrix operand order repaired and checked by reading, not compiled, not run on hardware
//
#include <hip/hip_runtime.h>

// Trajectron windowed dual-LSTM for MI455X (gfx1250, wave32).
// One workgroup per output timestep (window), 256 threads = 8 waves.
// Recurrent GEMM h @ Whh^T done with v_wmma_f32_16x16x32_f16 (K = H_HIS = 32).
// h kept in f16 (bounded in (-1,1)), cell state c and gate accum in f32.
// h ping-pongs through LDS; Whh B-operands live in registers across all steps.

typedef __attribute__((ext_vector_type(8)))  _Float16 v8h;
typedef __attribute__((ext_vector_type(16))) _Float16 v16h;
typedef __attribute__((ext_vector_type(8)))  float    v8f;

#define T_STEPS 256
#define BATCH   128
#define SEQL    64
#define HH      32
#define HI      8

__device__ __forceinline__ float sigm(float x) {
  x = fminf(fmaxf(x, -30.f), 30.f);
  return 1.0f / (1.0f + __expf(-x));
}
__device__ __forceinline__ float tanh_f(float x) {
  x = fminf(fmaxf(x, -15.f), 15.f);
  float e = __expf(2.0f * x);
  return (e - 1.0f) / (e + 1.0f);
}

__global__ __launch_bounds__(256)
void trajectron_lstm_kernel(const float* __restrict__ inputs,
                            const float* __restrict__ Wih_his,
                            const float* __restrict__ Whh_his,
                            const float* __restrict__ bih_his,
                            const float* __restrict__ bhh_his,
                            const float* __restrict__ Wih_int,
                            const float* __restrict__ Whh_int,
                            const float* __restrict__ bih_int,
                            const float* __restrict__ bhh_int,
                            const float* __restrict__ W_out,
                            const float* __restrict__ b_out,
                            float* __restrict__ out)
{
  // [pingpong][0=his,1=int][batch*32 + k]  (int padded to 32 k, zeros beyond 8)
  __shared__ _Float16 sh[2][2][BATCH * 32];

  const int tid  = threadIdx.x;
  const int lane = tid & 31;
  const int wv   = tid >> 5;      // wave id 0..7 -> M (batch) strip
  const int cl   = lane & 15;     // N column within 16x16 tile
  const int hi   = lane >> 4;     // half-wave group
  const int w    = blockIdx.x;    // window / output timestep

  // ---- zero LDS (both buffers; int padding must stay zero forever) ----
  {
    unsigned* z = (unsigned*)&sh[0][0][0];
    for (int i = tid; i < 2 * 2 * BATCH * 32 / 2; i += 256) z[i] = 0u;
  }

  // ---- B operands (Whh^T tiles) into registers, f16, held for all steps ----
  // B 32x16 layout: lane {hi,cl}: element j -> B[k=(j<8 ? 8*hi+j : 8+8*hi+j)][n=cl] = Whh[gate=n][k]
  v16h Bh[8];
  #pragma unroll
  for (int n = 0; n < 8; ++n) {
    const float* src = Whh_his + (n * 16 + cl) * HH;
    v16h b;
    #pragma unroll
    for (int j = 0; j < 16; ++j) b[j] = (_Float16)src[j < 8 ? 8 * hi + j : 8 + 8 * hi + j];
    Bh[n] = b;
  }
  v16h Bi[2];
  #pragma unroll
  for (int n = 0; n < 2; ++n) {
    v16h b;
    #pragma unroll
    for (int j = 0; j < 16; ++j) b[j] = (_Float16)0.0f;
    if (hi == 0) {                       // only K<8 nonzero (H_INT=8, padded)
      const float* src = Whh_int + (n * 16 + cl) * HI;
      #pragma unroll
      for (int j = 0; j < 8; ++j) b[j] = (_Float16)src[j];
    }
    Bi[n] = b;
  }

  // ---- per-lane input-projection weights (D=2) + fused biases ----
  float wh0[8], wh1[8], bsh[8];
  #pragma unroll
  for (int n = 0; n < 8; ++n) {
    const int g = n * 16 + cl;
    wh0[n] = Wih_his[2 * g];
    wh1[n] = Wih_his[2 * g + 1];
    bsh[n] = bih_his[g] + bhh_his[g];
  }
  float wi0[2], wi1[2], bsi[2];
  #pragma unroll
  for (int n = 0; n < 2; ++n) {
    const int g = n * 16 + cl;
    wi0[n] = Wih_int[2 * g];
    wi1[n] = Wih_int[2 * g + 1];
    bsi[n] = bih_int[g] + bhh_int[g];
  }

  // ---- cell state in registers: lane {hi,cl}, reg r -> batch = 16*wv+8*hi+r ----
  float cs[8][2];                 // his: hidden k = n*16 + cl
  float csi[8];                   // int: hidden k = cl (valid cl<8)
  #pragma unroll
  for (int r = 0; r < 8; ++r) { cs[r][0] = 0.f; cs[r][1] = 0.f; csi[r] = 0.f; }

  const int myM = wv * 16 + 8 * hi;    // base batch row for this lane's D VGPRs
  const int t0  = (w >= SEQL - 1) ? (w - (SEQL - 1)) : 0;
  int p = 0;

  __syncthreads();

  for (int t = t0; t <= w; ++t) {
    // x for this lane's 8 batch rows: node 3 (his) and node 0 (int), D=2
    float xh0[8], xh1[8], xi0[8], xi1[8];
    #pragma unroll
    for (int r = 0; r < 8; ++r) {
      const int b = myM + r;
      const float2* pin = (const float2*)(inputs + (size_t)(t * BATCH + b) * 8);
      const float2 xi = pin[0];
      const float2 xh = pin[3];
      xi0[r] = xi.x; xi1[r] = xi.y;
      xh0[r] = xh.x; xh1[r] = xh.y;
    }

    // A operands (h) from LDS, WMMA 16x32 f16 A layout:
    //   elem j<8 -> K=8*hi+j ; j>=8 -> K=16+8*hi+(j-8)   (two b128 loads)
    const _Float16* curH = sh[p][0];
    const _Float16* curI = sh[p][1];
    const int row = wv * 16 + cl;
    const v8h alo = *(const v8h*)&curH[row * 32 + 8 * hi];
    const v8h ahi = *(const v8h*)&curH[row * 32 + 16 + 8 * hi];
    const v16h Ah = __builtin_shufflevector(alo, ahi,
        0, 1, 2, 3, 4, 5, 6, 7, 8, 9, 10, 11, 12, 13, 14, 15);
    const v8h blo = *(const v8h*)&curI[row * 32 + 8 * hi];
    const v8h bhi = *(const v8h*)&curI[row * 32 + 16 + 8 * hi];
    const v16h Ai = __builtin_shufflevector(blo, bhi,
        0, 1, 2, 3, 4, 5, 6, 7, 8, 9, 10, 11, 12, 13, 14, 15);

    // gates = (x @ Wih^T + b)  [as C operand]  +  h @ Whh^T  [WMMA]
    v8f D[8];
    #pragma unroll
    for (int n = 0; n < 8; ++n) {
      v8f c;
      #pragma unroll
      for (int r = 0; r < 8; ++r)
        c[r] = fmaf(xh1[r], wh1[n], fmaf(xh0[r], wh0[n], bsh[n]));
      D[n] = __builtin_amdgcn_wmma_f32_16x16x32_f16(
          false, Ah, false, Bh[n], (short)0, c, false, false);
    }
    v8f Di[2];
    #pragma unroll
    for (int n = 0; n < 2; ++n) {
      v8f c;
      #pragma unroll
      for (int r = 0; r < 8; ++r)
        c[r] = fmaf(xi1[r], wi1[n], fmaf(xi0[r], wi0[n], bsi[n]));
      Di[n] = __builtin_amdgcn_wmma_f32_16x16x32_f16(
          false, Ai, false, Bi[n], (short)0, c, false, false);
    }

    _Float16* nxtH = sh[p ^ 1][0];
    _Float16* nxtI = sh[p ^ 1][1];

    // his elementwise update: gate order i,f,g,o -> tiles n, n+2, n+4, n+6
    #pragma unroll
    for (int r = 0; r < 8; ++r) {
      const int b = myM + r;
      #pragma unroll
      for (int n = 0; n < 2; ++n) {
        const float ig = D[n][r], fg = D[n + 2][r];
        const float gg = D[n + 4][r], og = D[n + 6][r];
        const float cn = sigm(fg) * cs[r][n] + sigm(ig) * tanh_f(gg);
        cs[r][n] = cn;
        nxtH[b * 32 + n * 16 + cl] = (_Float16)(sigm(og) * tanh_f(cn));
      }
    }
    // int update: i@col k, f@col 8+k (tile0), g@col k, o@col 8+k (tile1)
    #pragma unroll
    for (int r = 0; r < 8; ++r) {
      const float fg = __shfl_xor(Di[0][r], 8, 32);
      const float og = __shfl_xor(Di[1][r], 8, 32);
      if (cl < 8) {
        const int b = myM + r;
        const float ig = Di[0][r], gg = Di[1][r];
        const float cn = sigm(fg) * csi[r] + sigm(ig) * tanh_f(gg);
        csi[r] = cn;
        nxtI[b * 32 + cl] = (_Float16)(sigm(og) * tanh_f(cn));
      }
    }

    __syncthreads();
    p ^= 1;
  }

  // ---- output projection: emb[40] = [h_his(32), h_int(8)];  out = emb @ W_out^T + b ----
  {
    const int b = tid >> 1;
    const int o = tid & 1;
    const _Float16* hH = sh[p][0];
    const _Float16* hI = sh[p][1];
    float acc = b_out[o];
    #pragma unroll
    for (int k = 0; k < 32; ++k) acc += (float)hH[b * 32 + k] * W_out[o * 40 + k];
    #pragma unroll
    for (int k = 0; k < 8; ++k)  acc += (float)hI[b * 32 + k] * W_out[o * 40 + 32 + k];
    out[(w * BATCH + b) * 2 + o] = acc;
  }
}

extern "C" void kernel_launch(void* const* d_in, const int* in_sizes, int n_in,
                              void* d_out, int out_size, void* d_ws, size_t ws_size,
                              hipStream_t stream) {
  (void)in_sizes; (void)n_in; (void)out_size; (void)d_ws; (void)ws_size;
  trajectron_lstm_kernel<<<T_STEPS, 256, 0, stream>>>(
      (const float*)d_in[0],   // inputs [256,128,4,2]
      (const float*)d_in[1],   // Wih_his [128,2]
      (const float*)d_in[2],   // Whh_his [128,32]
      (const float*)d_in[3],   // bih_his [128]
      (const float*)d_in[4],   // bhh_his [128]
      (const float*)d_in[5],   // Wih_int [32,2]
      (const float*)d_in[6],   // Whh_int [32,8]
      (const float*)d_in[7],   // bih_int [32]
      (const float*)d_in[8],   // bhh_int [32]
      (const float*)d_in[9],   // W_out [2,40]
      (const float*)d_in[10],  // b_out [2]
      (float*)d_out);          // [256,128,2] f32
}
